// RelativeGlobalAttention_51848845197547
// MI455X (gfx1250) — compile-verified, matrix operand order checked by reading, not run on hardware
//
#include <hip/hip_runtime.h>
#include <hip/hip_bf16.h>

#define S_LEN   2048
#define D_MODEL 1024
#define N_HEADS 16
#define D_HEAD  64
#define BATCH   2

typedef _Float16 h8   __attribute__((ext_vector_type(8)));
typedef _Float16 v16h __attribute__((ext_vector_type(16)));
typedef float    v8f  __attribute__((ext_vector_type(8)));

#define WMMA_F16(a, b, c) \
  __builtin_amdgcn_wmma_f32_16x16x32_f16(false, (a), false, (b), (short)0, (c), false, false)

// ---------- CDNA5 async global->LDS copy (guarded; falls back to ld+ds_store) ----------
#if defined(__has_builtin)
#if __has_builtin(__builtin_amdgcn_global_load_async_to_lds_b128)
#define HAVE_ASYNC_CP 1
#endif
#endif
#ifndef HAVE_ASYNC_CP
#define HAVE_ASYNC_CP 0
#endif

typedef int v4i_t __attribute__((vector_size(16)));
typedef __attribute__((address_space(1))) v4i_t* g_v4i_p;   // global int4*
typedef __attribute__((address_space(3))) v4i_t* l_v4i_p;   // LDS int4*

static __device__ __forceinline__ void cp16_g2l(const _Float16* g, _Float16* l) {
#if HAVE_ASYNC_CP
  __builtin_amdgcn_global_load_async_to_lds_b128(
      (g_v4i_p)(unsigned long long)g,
      (l_v4i_p)(unsigned int)(unsigned long long)l, 0, 0);
#else
  *(h8*)l = *(const h8*)g;
#endif
}
static __device__ __forceinline__ void wait_async_le4() {
#if HAVE_ASYNC_CP
  asm volatile("s_wait_asynccnt 0x4" ::: "memory");
#endif
}
static __device__ __forceinline__ void wait_async_le0() {
#if HAVE_ASYNC_CP
  asm volatile("s_wait_asynccnt 0x0" ::: "memory");
#endif
}

// Load one lane's slice of a 16x32 (f16) A/B fragment.
// Layout per ISA 7.12.2: lanes 0-15 hold K[0..7]+K[16..23], lanes 16-31 hold K[8..15]+K[24..31].
static __device__ __forceinline__ v16h frag_ld(const _Float16* p, int k0) {
  h8 lo = *(const h8*)(p + k0);
  h8 hi = *(const h8*)(p + k0 + 16);
  return __builtin_shufflevector(lo, hi, 0,1,2,3,4,5,6,7,8,9,10,11,12,13,14,15);
}

// G = Qtile(16x64) @ Er[ebase .. ebase+15]^T  -> 16x16 f32 (2 WMMAs over K=64)
static __device__ __forceinline__ v8f er_gemm(v16h a0, v16h a1,
                                              const _Float16* __restrict__ Erh,
                                              int ebase, int nn, int k0) {
  int e = ebase + nn;
  e = e < 0 ? 0 : e;
  e = e > (S_LEN - 1) ? (S_LEN - 1) : e;   // clamp; clamped columns are never selected
  const _Float16* col = Erh + (size_t)e * D_HEAD;
  v8f c = {};
  c = WMMA_F16(a0, frag_ld(col, k0), c);
  c = WMMA_F16(a1, frag_ld(col + 32, k0), c);
  return c;
}

// ---------------- Kernel 1: f32 -> f16 cast ----------------
__global__ void cast_f32_f16(const float* __restrict__ in, _Float16* __restrict__ out, int n) {
  int i = blockIdx.x * blockDim.x + threadIdx.x;
  int stride = gridDim.x * blockDim.x;
  for (; i < n; i += stride) out[i] = (_Float16)in[i];
}

// ---------------- Kernel 2: QKV projection GEMM ----------------
// y = Xh @ W^T + b ; z = 0/1/2 -> Q/K/V. Q,K stored (b,h,s,dh); V stored transposed (b,h,dh,s).
// B panel (16 output features x K=1024) is staged once into LDS via async copies and
// shared by all 4 waves; A streams from global (L2-resident).
__global__ void __launch_bounds__(128)
qkv_gemm(const _Float16* __restrict__ Xh,
         const _Float16* __restrict__ Wqh, const _Float16* __restrict__ Wkh,
         const _Float16* __restrict__ Wvh,
         const float* __restrict__ bq, const float* __restrict__ bk, const float* __restrict__ bv,
         _Float16* __restrict__ Qh, _Float16* __restrict__ Kh, _Float16* __restrict__ Vth) {
  __shared__ __align__(16) _Float16 Bl[16][D_MODEL];   // 32 KB

  const int tid  = threadIdx.x;
  const int lane = tid & 31;
  const int wave = tid >> 5;
  const int nn   = lane & 15;
  const int k0   = ((lane >> 4) & 1) * 8;
  const int n0   = blockIdx.x * 16;
  const int row0 = (blockIdx.y * 4 + wave) * 16;
  const int which = blockIdx.z;

  const _Float16* W    = (which == 0) ? Wqh : (which == 1) ? Wkh : Wvh;
  const float*    bias = (which == 0) ? bq  : (which == 1) ? bk  : bv;

  // cooperative async B-panel load: 16 cols x 1024 halfs = 2048 x 16B chunks
#pragma unroll
  for (int r = 0; r < 16; ++r) {
    int c    = tid + r * 128;
    int col  = c >> 7;
    int kpos = (c & 127) << 3;
    cp16_g2l(W + (size_t)(n0 + col) * D_MODEL + kpos, &Bl[col][kpos]);
  }
  wait_async_le0();
  __syncthreads();

  const _Float16* arow = Xh + (size_t)(row0 + nn) * D_MODEL;  // A: row = lane&15
  const _Float16* bcol = &Bl[nn][0];                          // B col = staged W row

  v8f acc = {};
#pragma unroll 8
  for (int kk = 0; kk < D_MODEL; kk += 32) {
    acc = WMMA_F16(frag_ld(arow + kk, k0), frag_ld(bcol + kk, k0), acc);
  }

  const int   o   = n0 + nn;
  const float bb_ = bias[o];
  const int   h   = o >> 6, dh = o & 63;
  const int   hiRow = ((lane >> 4) & 1) * 8;
#pragma unroll
  for (int g = 0; g < 8; ++g) {
    int r = row0 + g + hiRow;          // C layout: lane holds col o, row g(+8)
    int bb = r >> 11;                  // / S_LEN
    int s  = r & (S_LEN - 1);
    _Float16 val = (_Float16)(acc[g] + bb_);
    size_t head = (size_t)(bb * N_HEADS + h);
    if (which == 2)      Vth[(head * D_HEAD + dh) * S_LEN + s] = val;
    else if (which == 0) Qh[(head * S_LEN + s) * D_HEAD + dh]  = val;
    else                 Kh[(head * S_LEN + s) * D_HEAD + dh]  = val;
  }
}

// ---------------- Kernel 3: flash attention with relative skew ----------------
// K/V slabs (32 keys) are double-buffered in LDS via async copies, shared by all 4 waves;
// prefetch of slab n+1 overlaps compute on slab n (s_wait_asynccnt <= 4).
static __device__ __forceinline__ void slab_load(const _Float16* __restrict__ Kbase,
                                                 const _Float16* __restrict__ Vbase,
                                                 int t0,
                                                 _Float16 (*Kl)[64], _Float16 (*Vl)[32],
                                                 int tid) {
#pragma unroll
  for (int r = 0; r < 2; ++r) {
    int c = tid + r * 128;                 // 0..255
    int krow = c >> 3, kcol = (c & 7) << 3;   // 32 rows x 64 halfs
    cp16_g2l(Kbase + (size_t)(t0 + krow) * D_HEAD + kcol, &Kl[krow][kcol]);
    int vrow = c >> 2, vcol = (c & 3) << 3;   // 64 rows x 32 halfs
    cp16_g2l(Vbase + (size_t)vrow * S_LEN + t0 + vcol, &Vl[vrow][vcol]);
  }
}

__global__ void __launch_bounds__(128)
attn_kernel(const _Float16* __restrict__ Qh, const _Float16* __restrict__ Kh,
            const _Float16* __restrict__ Vt, const _Float16* __restrict__ Erh,
            float* __restrict__ out) {
  __shared__ __align__(16) _Float16 Kl[2][32][64];   // 8 KB double-buffered K slab
  __shared__ __align__(16) _Float16 Vl[2][64][32];   // 8 KB double-buffered V slab (dh,t)
  __shared__ __align__(16) _Float16 plds[4][16][32]; // per-wave P transpose staging

  const int tid  = threadIdx.x;
  const int lane = tid & 31;
  const int wave = tid >> 5;
  const int hiH  = (lane >> 4) & 1;
  const int k0   = hiH * 8;
  const int nn   = lane & 15;

  const int h  = blockIdx.y;
  const int b  = blockIdx.z;
  const int s0 = blockIdx.x * 64 + wave * 16;

  const size_t head = (size_t)(b * N_HEADS + h) * (size_t)S_LEN * D_HEAD;
  const _Float16* Qbase = Qh + head;
  const _Float16* Kbase = Kh + head;
  const _Float16* Vbase = Vt + head;   // (dh, t) layout

  // warm the Er diagonal band this block touches first
  {
    int e = S_LEN - 16 - s0 + nn;
    e = e < 0 ? 0 : e;
    __builtin_prefetch(Erh + (size_t)e * D_HEAD, 0, 1);
  }

  const _Float16* qrow = Qbase + (size_t)(s0 + nn) * D_HEAD;
  int sshift = s0 + nn + 1; if (sshift > S_LEN - 1) sshift = S_LEN - 1;
  const _Float16* qsrow = Qbase + (size_t)sshift * D_HEAD;  // rows s+1 (skew wrap term)

  const v16h qa0 = frag_ld(qrow, k0);        // Q  d 0..31
  const v16h qa1 = frag_ld(qrow + 32, k0);   // Q  d 32..63
  const v16h qs0 = frag_ld(qsrow, k0);       // Qs d 0..31
  const v16h qs1 = frag_ld(qsrow + 32, k0);  // Qs d 32..63

  float rowmax[8], rowsum[8];
  v8f o0 = {}, o1 = {}, o2 = {}, o3 = {};
#pragma unroll
  for (int g = 0; g < 8; ++g) { rowmax[g] = -3.0e38f; rowsum[g] = 0.f; }

  slab_load(Kbase, Vbase, 0, Kl[0], Vl[0], tid);   // preload slab 0

  for (int t0 = 0; t0 < S_LEN; t0 += 32) {
    const int buf = (t0 >> 5) & 1;
    if (t0 + 32 < S_LEN) {
      slab_load(Kbase, Vbase, t0 + 32, Kl[buf ^ 1], Vl[buf ^ 1], tid);
      wait_async_le4();                // current slab landed; prefetch stays in flight
    } else {
      wait_async_le0();
    }
    __syncthreads();

    float sc[2][8];
    float tmax[8];
#pragma unroll
    for (int g = 0; g < 8; ++g) tmax[g] = -3.0e38f;

#pragma unroll
    for (int sub = 0; sub < 2; ++sub) {
      const int tt = t0 + sub * 16;

      // ---- QK^T 16x16 tile (K = 64) from LDS ----
      const _Float16* kcol = &Kl[buf][sub * 16 + nn][0];
      v8f c = {};
      c = WMMA_F16(qa0, frag_ld(kcol, k0), c);
      c = WMMA_F16(qa1, frag_ld(kcol + 32, k0), c);

      // ---- Srel tile via Er-block WMMAs + Toeplitz lane shift ----
      const int d0 = tt - s0;
      float srel[8];
#pragma unroll
      for (int g = 0; g < 8; ++g) srel[g] = 0.f;

      if (d0 <= 15) {  // in-band: Srel[s,t] = q[s].Er[S-1+t-s],  t<=s
        v8f g1 = er_gemm(qa0, qa1, Erh, S_LEN - 16 + d0, nn, k0);
        v8f g2 = er_gemm(qa0, qa1, Erh, S_LEN + d0,      nn, k0);
#pragma unroll
        for (int g = 0; g < 8; ++g) {
          int mm = g + 8 * hiH;
          int jm = nn - mm;                     // j - i
          int dd = d0 + jm;                     // t - s
          int srcc = (jm >= 1) ? (jm - 1) : (jm + 15);
          int srcl = (lane & 16) | srcc;
          float v1 = __shfl(g1[g], srcl, 32);
          float v2 = __shfl(g2[g], srcl, 32);
          float v  = (jm >= 1) ? v2 : v1;
          if (dd <= 0) srel[g] = v;
        }
      }
      if (d0 >= -13) { // wrap artifact: Srel[s,t] = q[s+1].Er[t-s-2],  t>=s+2
        v8f gj1 = er_gemm(qs0, qs1, Erh, d0 - 17, nn, k0);
        v8f gj2 = er_gemm(qs0, qs1, Erh, d0 - 1,  nn, k0);
#pragma unroll
        for (int g = 0; g < 8; ++g) {
          int mm = g + 8 * hiH;
          int jm = nn - mm;
          int dd = d0 + jm;
          int srcc = (jm >= 1) ? (jm - 1) : (jm + 15);
          int srcl = (lane & 16) | srcc;
          float v1 = __shfl(gj1[g], srcl, 32);
          float v2 = __shfl(gj2[g], srcl, 32);
          float v  = (jm >= 1) ? v2 : v1;
          if (dd >= 2) srel[g] = v;
        }
      }
      // (dd == 1 stays exactly 0, matching the skew's zero column)

#pragma unroll
      for (int g = 0; g < 8; ++g) {
        float sv = (c[g] + srel[g]) * 0.125f;   // / sqrt(D_HEAD)
        sc[sub][g] = sv;
        tmax[g] = fmaxf(tmax[g], sv);
      }
    }

    // ---- online softmax across this 32-key slab ----
#pragma unroll
    for (int g = 0; g < 8; ++g) {
      float tm = tmax[g];
#pragma unroll
      for (int x = 1; x < 16; x <<= 1) tm = fmaxf(tm, __shfl_xor(tm, x, 32));
      float nm = fmaxf(rowmax[g], tm);
      float al = __expf(rowmax[g] - nm);
      rowmax[g] = nm;
      float p0 = __expf(sc[0][g] - nm);
      float p1 = __expf(sc[1][g] - nm);
      float ts = p0 + p1;
#pragma unroll
      for (int x = 1; x < 16; x <<= 1) ts += __shfl_xor(ts, x, 32);
      rowsum[g] = rowsum[g] * al + ts;
      o0[g] *= al; o1[g] *= al; o2[g] *= al; o3[g] *= al;
      int mm = g + 8 * hiH;
      plds[wave][mm][nn]      = (_Float16)p0;   // C-layout scatter into LDS
      plds[wave][mm][16 + nn] = (_Float16)p1;
    }

    // read back P as A-fragment (row = lane&15, K = 32 keys)
    v16h pa = frag_ld(&plds[wave][nn][0], k0);

    // ---- P @ V : 4 WMMAs over the 64 head dims (K = 32), V from LDS ----
    const _Float16* vrow_ = &Vl[buf][nn][0];              // row stride 32 halfs
    o0 = WMMA_F16(pa, frag_ld(vrow_,            k0), o0);
    o1 = WMMA_F16(pa, frag_ld(vrow_ + 16 * 32,  k0), o1);
    o2 = WMMA_F16(pa, frag_ld(vrow_ + 32 * 32,  k0), o2);
    o3 = WMMA_F16(pa, frag_ld(vrow_ + 48 * 32,  k0), o3);

    __syncthreads();   // slab buffers free for the next prefetch
  }

  // ---- normalize and write (B, S, D_MODEL) f32 ----
#pragma unroll
  for (int g = 0; g < 8; ++g) {
    int s = s0 + g + 8 * hiH;
    float inv = 1.0f / rowsum[g];
    size_t base = ((size_t)(b * S_LEN + s)) * D_MODEL + h * D_HEAD + nn;
    out[base +  0] = o0[g] * inv;
    out[base + 16] = o1[g] * inv;
    out[base + 32] = o2[g] * inv;
    out[base + 48] = o3[g] * inv;
  }
}

extern "C" void kernel_launch(void* const* d_in, const int* in_sizes, int n_in,
                              void* d_out, int out_size, void* d_ws, size_t ws_size,
                              hipStream_t stream) {
  (void)in_sizes; (void)n_in; (void)out_size; (void)ws_size;
  const float* x  = (const float*)d_in[0];
  const float* Wq = (const float*)d_in[1];
  const float* bq = (const float*)d_in[2];
  const float* Wk = (const float*)d_in[3];
  const float* bk = (const float*)d_in[4];
  const float* Wv = (const float*)d_in[5];
  const float* bv = (const float*)d_in[6];
  const float* Er = (const float*)d_in[7];

  char* ws = (char*)d_ws;
  const size_t xhN = (size_t)BATCH * S_LEN * D_MODEL;        // 4,194,304
  const size_t wN  = (size_t)D_MODEL * D_MODEL;              // 1,048,576
  const size_t erN = (size_t)S_LEN * D_HEAD;                 // 131,072

  _Float16* Xh  = (_Float16*)(ws);
  _Float16* Wqh = (_Float16*)(ws + 2 * xhN);
  _Float16* Wkh = Wqh + wN;
  _Float16* Wvh = Wkh + wN;
  _Float16* Erh = Wvh + wN;
  _Float16* Qh  = Erh + erN;
  _Float16* Kh  = Qh + xhN;
  _Float16* Vth = Kh + xhN;

  cast_f32_f16<<<2048, 256, 0, stream>>>(x,  Xh,  (int)xhN);
  cast_f32_f16<<<512,  256, 0, stream>>>(Wq, Wqh, (int)wN);
  cast_f32_f16<<<512,  256, 0, stream>>>(Wk, Wkh, (int)wN);
  cast_f32_f16<<<512,  256, 0, stream>>>(Wv, Wvh, (int)wN);
  cast_f32_f16<<<64,   256, 0, stream>>>(Er, Erh, (int)erN);

  // QKV: M = 4096 rows, N = 1024 feats, K = 1024; z selects Q/K/V
  qkv_gemm<<<dim3(D_MODEL / 16, (BATCH * S_LEN) / 64, 3), 128, 0, stream>>>(
      Xh, Wqh, Wkh, Wvh, bq, bk, bv, Qh, Kh, Vth);

  // attention: 64 query rows per block (4 waves x 16)
  attn_kernel<<<dim3(S_LEN / 64, N_HEADS, BATCH), 128, 0, stream>>>(
      Qh, Kh, Vth, Erh, (float*)d_out);
}
